// SwinTransformerBlock_56891136803094
// MI455X (gfx1250) — compile-verified, matrix operand order checked by reading, not run on hardware
//
#include <hip/hip_runtime.h>
#include <hip/hip_bf16.h>
#include <math.h>

typedef __attribute__((ext_vector_type(16))) _Float16 v16h;
typedef __attribute__((ext_vector_type(8)))  float    v8f;

#define Bn 4
#define Hh 192
#define Ww 192
#define Cc 192
#define Ll (Hh*Ww)                 // 36864
#define BLC ((size_t)Bn*Ll*Cc)     // 28,311,552
#define NHh 6
#define HD 32
#define NWIN (Bn*576)              // 2304 windows
#define EPSF 1e-5f

// k-offset of fragment pair h (0..7) within a 32-wide K chunk, per ISA 16-bit A/B layout:
// pairs at {0,2,4,6,16,18,20,22} + 8*hi
__device__ __forceinline__ int kpair(int h) { return ((h & 4) << 2) + ((h & 3) << 1); }

// ---------------- weight packing into WMMA B-fragment order ----------------
// Bp[(kc*Nt+nt)*512 + lane*16 + e] = W[k, n] (f16), matching v16h lane layout.
__global__ void pack_b(const float* __restrict__ W, _Float16* __restrict__ Bp, int K, int N) {
  int idx = blockIdx.x * 256 + threadIdx.x;
  int total = (K >> 5) * (N >> 4) * 512;
  if (idx >= total) return;
  int e = idx & 15, lane = (idx >> 4) & 31, tile = idx >> 9;
  int Nt = N >> 4;
  int kc = tile / Nt, nt = tile % Nt;
  int v = e >> 1, sub = e & 1;
  int k = (kc << 5) + kpair(v) + ((lane >> 4) << 3) + sub;
  int n = (nt << 4) + (lane & 15);
  Bp[idx] = (_Float16)W[(size_t)k * N + n];
}

// ---------------- LayerNorm (row of 192) -> f16 ----------------
__global__ __launch_bounds__(192) void ln_kernel(const float* __restrict__ X,
                                                 const float* __restrict__ g,
                                                 const float* __restrict__ bta,
                                                 _Float16* __restrict__ Y) {
  size_t row = blockIdx.x;
  int t = threadIdx.x;
  float v = X[row * Cc + t];
  float s = v, ss = v * v;
  for (int o = 16; o > 0; o >>= 1) { s += __shfl_down(s, o, 32); ss += __shfl_down(ss, o, 32); }
  __shared__ float ps[6], pss[6];
  int lane = t & 31, w = t >> 5;
  if (lane == 0) { ps[w] = s; pss[w] = ss; }
  __syncthreads();
  float S = 0.f, SS = 0.f;
  for (int i = 0; i < 6; ++i) { S += ps[i]; SS += pss[i]; }
  float mu = S * (1.f / Cc);
  float var = SS * (1.f / Cc) - mu * mu;
  float inv = rsqrtf(var + EPSF);
  Y[row * Cc + t] = (_Float16)((v - mu) * inv * g[t] + bta[t]);
}

__device__ __forceinline__ float apply_act(float v, int act) {
  if (act == 1) return 0.5f * v * (1.0f + erff(v * 0.70710678118f));
  if (act == 2) return fmaxf(v, 0.f);
  return v;
}

// ---------------- generic WMMA GEMM, register-blocked 32x64 per wave ----------------
// C = act(A@W + bias) [+= existing]. A: MxK row-major f16; Bp packed.
// Requires M % 32 == 0, N % 64 == 0, K % 32 == 0.
__global__ __launch_bounds__(256) void gemm_wmma(const _Float16* __restrict__ A,
                                                 const _Float16* __restrict__ Bp,
                                                 const float* __restrict__ bias,
                                                 float* __restrict__ Cf, _Float16* __restrict__ Ch,
                                                 int M, int N, int K, int act, int accum) {
  int Ng = N >> 6, Nt = N >> 4;
  int wid = blockIdx.x * 8 + (threadIdx.x >> 5);
  if (wid >= (M >> 5) * Ng) return;            // wave-uniform exit (EXEC stays all-1 for WMMA)
  int mg = wid / Ng, ng = wid % Ng;
  int l = threadIdx.x & 31, hi = l >> 4, ln = l & 15;
  size_t m0 = (size_t)(mg << 5) + ln;
  v8f acc[2][4] = {};
  int KC = K >> 5;
  for (int kc = 0; kc < KC; ++kc) {
    v16h a0, a1;
#pragma unroll
    for (int h = 0; h < 8; ++h) {
      const _Float16* p0 = A + m0 * K + (kc << 5) + kpair(h) + hi * 8;
      a0[2 * h] = p0[0]; a0[2 * h + 1] = p0[1];
      const _Float16* p1 = p0 + (size_t)16 * K;
      a1[2 * h] = p1[0]; a1[2 * h + 1] = p1[1];
    }
    const _Float16* bbase = Bp + (((size_t)kc * Nt + (ng << 2)) << 9) + (l << 4);
#pragma unroll
    for (int j = 0; j < 4; ++j) {
      v16h b = *(const v16h*)(bbase + (j << 9));
      acc[0][j] = __builtin_amdgcn_wmma_f32_16x16x32_f16(false, a0, false, b, (short)0, acc[0][j], false, false);
      acc[1][j] = __builtin_amdgcn_wmma_f32_16x16x32_f16(false, a1, false, b, (short)0, acc[1][j], false, false);
    }
  }
#pragma unroll
  for (int i = 0; i < 2; ++i)
#pragma unroll
    for (int j = 0; j < 4; ++j) {
      int n = ((ng << 2) + j) * 16 + ln;
      float bv = bias[n];
#pragma unroll
      for (int r = 0; r < 8; ++r) {
        size_t mr = (size_t)(mg << 5) + i * 16 + r + hi * 8;
        float v = apply_act(acc[i][j][r] + bv, act);
        size_t o = mr * N + n;
        if (Cf) { if (accum) v += Cf[o]; Cf[o] = v; }
        if (Ch) Ch[o] = (_Float16)v;
      }
    }
}

// ---------------- implicit-GEMM 3x3 SAME conv via WMMA, 32x64 per wave ----------------
// src: (Bn,H,W,cin) f16; weights packed from (3,3,cin,N) flat (K=9*cin, N).
__global__ __launch_bounds__(256) void conv_wmma(const _Float16* __restrict__ src,
                                                 const _Float16* __restrict__ Bp,
                                                 const float* __restrict__ bias,
                                                 float* __restrict__ Cf, _Float16* __restrict__ Ch,
                                                 int cin, int N, int act) {
  int Ng = (N >= 64) ? (N >> 6) : 1, Nt = N >> 4;
  int jmax = (N >= 64) ? 4 : (N >> 4);
  int wid = blockIdx.x * 8 + (threadIdx.x >> 5);
  if (wid >= ((Bn * Ll) >> 5) * Ng) return;
  int mg = wid / Ng, ng = wid % Ng;
  int l = threadIdx.x & 31, hi = l >> 4, ln = l & 15;
  int m0 = (mg << 5) + ln, m1 = m0 + 16;
  int b0 = m0 / Ll, yx0 = m0 % Ll, y0 = yx0 / Ww, x0 = yx0 % Ww;
  int b1 = m1 / Ll, yx1 = m1 % Ll, y1 = yx1 / Ww, x1 = yx1 % Ww;
  v8f acc[2][4] = {};
  int KC = (9 * cin) >> 5;
  for (int kc = 0; kc < KC; ++kc) {
    int tap = (kc << 5) / cin;                // whole chunk is inside one tap (cin % 32 == 0)
    int dy = tap / 3 - 1, dx = tap % 3 - 1;
    int cib = (kc << 5) - tap * cin;
    int ya0 = y0 + dy, xa0 = x0 + dx, ya1 = y1 + dy, xa1 = x1 + dx;
    bool in0 = (ya0 >= 0) & (ya0 < Hh) & (xa0 >= 0) & (xa0 < Ww);
    bool in1 = (ya1 >= 0) & (ya1 < Hh) & (xa1 >= 0) & (xa1 < Ww);
    const _Float16* p0 = src + (((size_t)(b0 * Hh + ya0) * Ww + xa0) * cin);
    const _Float16* p1 = src + (((size_t)(b1 * Hh + ya1) * Ww + xa1) * cin);
    v16h a0, a1;
#pragma unroll
    for (int h = 0; h < 8; ++h) {
      int ci = cib + kpair(h) + hi * 8;
      _Float16 z = (_Float16)0.f;
      a0[2 * h]     = in0 ? p0[ci]     : z;
      a0[2 * h + 1] = in0 ? p0[ci + 1] : z;
      a1[2 * h]     = in1 ? p1[ci]     : z;
      a1[2 * h + 1] = in1 ? p1[ci + 1] : z;
    }
    const _Float16* bbase = Bp + (((size_t)kc * Nt + ng * jmax) << 9) + (l << 4);
    for (int j = 0; j < jmax; ++j) {
      v16h bf = *(const v16h*)(bbase + (j << 9));
      acc[0][j] = __builtin_amdgcn_wmma_f32_16x16x32_f16(false, a0, false, bf, (short)0, acc[0][j], false, false);
      acc[1][j] = __builtin_amdgcn_wmma_f32_16x16x32_f16(false, a1, false, bf, (short)0, acc[1][j], false, false);
    }
  }
  for (int i = 0; i < 2; ++i)
    for (int j = 0; j < jmax; ++j) {
      int n = (ng * jmax + j) * 16 + ln;
      float bv = bias[n];
#pragma unroll
      for (int r = 0; r < 8; ++r) {
        size_t mr = (size_t)(mg << 5) + i * 16 + r + hi * 8;
        float v = apply_act(acc[i][j][r] + bv, act);
        size_t o = mr * N + n;
        if (Cf) Cf[o] = v;
        if (Ch) Ch[o] = (_Float16)v;
      }
    }
}

// ---------------- shifted window partition (roll -4,-4) ----------------
__global__ void win_part(const _Float16* __restrict__ xn, _Float16* __restrict__ wins) {
  size_t i = (size_t)blockIdx.x * 256 + threadIdx.x;
  int c = i % Cc; size_t wp = i / Cc;
  int p = wp & 63; int win = wp >> 6;
  int b = win / 576, wi = win % 576;
  int wy = wi / 24, wx = wi % 24, py = p >> 3, px = p & 7;
  int ys = (wy * 8 + py + 4) % Hh, xs = (wx * 8 + px + 4) % Ww;
  wins[i] = xn[(((size_t)(b * Hh + ys)) * Ww + xs) * Cc + c];
}

// ---------------- window attention: S=QK^T(+bias+mask), softmax, O=PV ----------------
__global__ __launch_bounds__(256) void attn_kernel(const _Float16* __restrict__ qkv,
                                                   const float* __restrict__ rpb,
                                                   _Float16* __restrict__ aout) {
  __shared__ float    Sm[64 * 64];
  __shared__ _Float16 Pm[64 * 64];
  int wh = blockIdx.x;
  int win = wh / NHh, head = wh % NHh;
  int wi = win % 576, wy = wi / 24, wx = wi % 24;
  int t = threadIdx.x, w = t >> 5, l = t & 31, hi = l >> 4, ln = l & 15;
  size_t base = (size_t)win * 64 * 576;
  // ---- S = (q*scale) k^T : 16 tiles, 2 per wave ----
  for (int tt = w * 2; tt < w * 2 + 2; ++tt) {
    int mt = tt >> 2, nt = tt & 3;
    v16h a, bf;
#pragma unroll
    for (int h = 0; h < 8; ++h) {
      int d = kpair(h) + hi * 8;
      const _Float16* pq = qkv + base + (size_t)(mt * 16 + ln) * 576 + head * HD + d;
      a[2 * h] = pq[0]; a[2 * h + 1] = pq[1];
      const _Float16* pk = qkv + base + (size_t)(nt * 16 + ln) * 576 + 192 + head * HD + d;
      bf[2 * h] = pk[0]; bf[2 * h + 1] = pk[1];
    }
    v8f acc = {};
    acc = __builtin_amdgcn_wmma_f32_16x16x32_f16(false, a, false, bf, (short)0, acc, false, false);
#pragma unroll
    for (int r = 0; r < 8; ++r) {
      int pm = mt * 16 + r + hi * 8, pn = nt * 16 + ln;
      int yi = pm >> 3, xi = pm & 7, yj = pn >> 3, xj = pn & 7;
      float bias = rpb[((yi - yj + 7) * 15 + (xi - xj + 7)) * NHh + head];
      int ym = wy * 8 + yi, xm = wx * 8 + xi, yn = wy * 8 + yj, xn2 = wx * 8 + xj;
      int cm = ((ym < 184) ? 0 : (ym < 188 ? 1 : 2)) * 3 + ((xm < 184) ? 0 : (xm < 188 ? 1 : 2));
      int cn = ((yn < 184) ? 0 : (yn < 188 ? 1 : 2)) * 3 + ((xn2 < 184) ? 0 : (xn2 < 188 ? 1 : 2));
      Sm[pm * 64 + pn] = acc[r] * 0.17677669529663687f + bias + ((cm == cn) ? 0.f : -100.f);
    }
  }
  __syncthreads();
  // ---- softmax: 4 lanes per row, 16 elems each ----
  {
    int row = t >> 2, seg = t & 3, b0 = row * 64 + seg * 16;
    float mx = -1e30f;
#pragma unroll
    for (int i = 0; i < 16; ++i) mx = fmaxf(mx, Sm[b0 + i]);
    mx = fmaxf(mx, __shfl_xor(mx, 1, 4));
    mx = fmaxf(mx, __shfl_xor(mx, 2, 4));
    float s = 0.f;
#pragma unroll
    for (int i = 0; i < 16; ++i) { float e = __expf(Sm[b0 + i] - mx); s += e; Pm[b0 + i] = (_Float16)e; }
    s += __shfl_xor(s, 1, 4);
    s += __shfl_xor(s, 2, 4);
    float inv = 1.f / s;
#pragma unroll
    for (int i = 0; i < 16; ++i) Pm[b0 + i] = (_Float16)((float)Pm[b0 + i] * inv);
  }
  __syncthreads();
  // ---- O = P V : 8 tiles (4x2), 1 per wave ----
  {
    int mt = w >> 1, nt = w & 1;
    v8f acc = {};
    for (int kc = 0; kc < 2; ++kc) {
      v16h a, bf;
#pragma unroll
      for (int h = 0; h < 8; ++h) {
        const _Float16* p = Pm + (mt * 16 + ln) * 64 + kc * 32 + kpair(h) + hi * 8;
        a[2 * h] = p[0]; a[2 * h + 1] = p[1];
      }
#pragma unroll
      for (int h = 0; h < 16; ++h) {
        int k = kc * 32 + kpair(h >> 1) + hi * 8 + (h & 1);
        bf[h] = qkv[base + (size_t)k * 576 + 384 + head * HD + nt * 16 + ln];
      }
      acc = __builtin_amdgcn_wmma_f32_16x16x32_f16(false, a, false, bf, (short)0, acc, false, false);
    }
#pragma unroll
    for (int r = 0; r < 8; ++r) {
      int pm = mt * 16 + r + hi * 8;
      aout[((size_t)win * 64 + pm) * Cc + head * HD + nt * 16 + ln] = (_Float16)acc[r];
    }
  }
}

// ---------------- CAB global pool ----------------
__global__ __launch_bounds__(256) void cab_pool(const float* __restrict__ cv, float* __restrict__ pool) {
  int b = blockIdx.x / Cc, c = blockIdx.x % Cc;
  float s = 0.f;
  for (int k = threadIdx.x; k < Ll; k += 256) s += cv[((size_t)b * Ll + k) * Cc + c];
  __shared__ float sd[256];
  sd[threadIdx.x] = s; __syncthreads();
  for (int o = 128; o > 0; o >>= 1) { if (threadIdx.x < o) sd[threadIdx.x] += sd[threadIdx.x + o]; __syncthreads(); }
  if (threadIdx.x == 0) pool[blockIdx.x] = sd[0] * (1.f / Ll);
}

// ---------------- channel attention MLP (tiny) ----------------
__global__ __launch_bounds__(256) void ca_kernel(const float* __restrict__ pool,
                                                 const float* __restrict__ f1, const float* __restrict__ b1,
                                                 const float* __restrict__ f2, const float* __restrict__ b2,
                                                 float* __restrict__ ya) {
  __shared__ float tj[Bn][NHh];
  int t = threadIdx.x;
  if (t < Bn * NHh) {
    int b = t / NHh, j = t % NHh;
    float s = b1[j];
    for (int c = 0; c < Cc; ++c) s += pool[b * Cc + c] * f1[c * NHh + j];
    tj[b][j] = fmaxf(s, 0.f);
  }
  __syncthreads();
  for (int i = t; i < Bn * Cc; i += 256) {
    int b = i / Cc, c = i % Cc;
    float s = b2[c];
    for (int j = 0; j < NHh; ++j) s += tj[b][j] * f2[j * Cc + c];
    ya[i] = 1.f / (1.f + __expf(-s));
  }
}

// ---------------- HIN instance-norm stats (first 96 ch) ----------------
__global__ __launch_bounds__(256) void hin_stats(const float* __restrict__ hv1,
                                                 float* __restrict__ hm, float* __restrict__ hv) {
  int b = blockIdx.x / 96, c = blockIdx.x % 96;
  float s = 0.f, ss = 0.f;
  for (int k = threadIdx.x; k < Ll; k += 256) {
    float v = hv1[((size_t)b * Ll + k) * Cc + c];
    s += v; ss += v * v;
  }
  __shared__ float sd[256], sd2[256];
  sd[threadIdx.x] = s; sd2[threadIdx.x] = ss; __syncthreads();
  for (int o = 128; o > 0; o >>= 1) {
    if (threadIdx.x < o) { sd[threadIdx.x] += sd[threadIdx.x + o]; sd2[threadIdx.x] += sd2[threadIdx.x + o]; }
    __syncthreads();
  }
  if (threadIdx.x == 0) {
    float mu = sd[0] * (1.f / Ll);
    hm[blockIdx.x] = mu;
    hv[blockIdx.x] = sd2[0] * (1.f / Ll) - mu * mu;
  }
}

// ---------------- HIN norm + relu -> f16 ----------------
__global__ void hin_norm(const float* __restrict__ hv1, const float* __restrict__ hm,
                         const float* __restrict__ hva, const float* __restrict__ g,
                         const float* __restrict__ bb, _Float16* __restrict__ out) {
  size_t i = (size_t)blockIdx.x * 256 + threadIdx.x;
  int c = i % Cc;
  int b = (int)(i / ((size_t)Ll * Cc));
  float v = hv1[i];
  if (c < 96) v = (v - hm[b * 96 + c]) * rsqrtf(hva[b * 96 + c] + EPSF) * g[c] + bb[c];
  out[i] = (_Float16)fmaxf(v, 0.f);
}

// ---------------- residual combine (with window-reverse + roll back) ----------------
__global__ void combine_kernel(const float* __restrict__ x, const float* __restrict__ proj,
                               const float* __restrict__ cv, const float* __restrict__ ya,
                               const float* __restrict__ hin, float* __restrict__ x2) {
  size_t i = (size_t)blockIdx.x * 256 + threadIdx.x;
  int c = i % Cc; size_t bl = i / Cc;
  int lidx = (int)(bl % Ll); int b = (int)(bl / Ll);
  int y = lidx / Ww, xx = lidx % Ww;
  int Y = (y + Hh - 4) % Hh, X = (xx + Ww - 4) % Ww;
  size_t wp = ((size_t)(b * 576 + (Y >> 3) * 24 + (X >> 3)) << 6) + ((Y & 7) << 3) + (X & 7);
  x2[i] = x[i] + proj[wp * Cc + c] + 0.01f * (cv[i] * ya[b * Cc + c] + hin[i]);
}

// =====================================================================
extern "C" void kernel_launch(void* const* d_in, const int* in_sizes, int n_in,
                              void* d_out, int out_size, void* d_ws, size_t ws_size,
                              hipStream_t stream) {
  const float* x       = (const float*)d_in[0];
  const float* n1g     = (const float*)d_in[1];
  const float* n1b     = (const float*)d_in[2];
  const float* qkv_w   = (const float*)d_in[3];
  const float* qkv_b   = (const float*)d_in[4];
  const float* rpb     = (const float*)d_in[5];
  const float* proj_w  = (const float*)d_in[6];
  const float* proj_b  = (const float*)d_in[7];
  const float* n2g     = (const float*)d_in[8];
  const float* n2b     = (const float*)d_in[9];
  const float* fc1_w   = (const float*)d_in[10];
  const float* fc1_b   = (const float*)d_in[11];
  const float* fc2_w   = (const float*)d_in[12];
  const float* fc2_b   = (const float*)d_in[13];
  const float* cab1_w  = (const float*)d_in[14];
  const float* cab1_b  = (const float*)d_in[15];
  const float* cab2_w  = (const float*)d_in[16];
  const float* cab2_b  = (const float*)d_in[17];
  const float* caf1_w  = (const float*)d_in[18];
  const float* caf1_b  = (const float*)d_in[19];
  const float* caf2_w  = (const float*)d_in[20];
  const float* caf2_b  = (const float*)d_in[21];
  const float* hin1_w  = (const float*)d_in[22];
  const float* hin1_b  = (const float*)d_in[23];
  const float* hing    = (const float*)d_in[24];
  const float* hinb    = (const float*)d_in[25];
  const float* hin2_w  = (const float*)d_in[26];
  const float* hin2_b  = (const float*)d_in[27];
  const float* hid_w   = (const float*)d_in[28];
  const float* hid_b   = (const float*)d_in[29];

  char* W = (char*)d_ws;
  size_t off = 0;
  auto alloc = [&](size_t bytes) { size_t o = off; off += (bytes + 255) & ~(size_t)255; return (void*)(W + o); };

  _Float16* xn16   = (_Float16*)alloc(BLC * 2);
  _Float16* wins16 = (_Float16*)alloc(BLC * 2);
  _Float16* qkv16  = (_Float16*)alloc(BLC * 3 * 2);
  _Float16* aout16 = (_Float16*)alloc(BLC * 2);
  float*    proj32 = (float*)   alloc(BLC * 4);
  _Float16* cv1_16 = (_Float16*)alloc((size_t)Bn * Ll * 64 * 2);
  float*    cv32   = (float*)   alloc(BLC * 4);
  float*    pool32 = (float*)   alloc(Bn * Cc * 4);
  float*    ya32   = (float*)   alloc(Bn * Cc * 4);
  float*    hv1_32 = (float*)   alloc(BLC * 4);
  float*    hm32   = (float*)   alloc(Bn * 96 * 4);
  float*    hva32  = (float*)   alloc(Bn * 96 * 4);
  _Float16* hv16   = (_Float16*)alloc(BLC * 2);
  float*    hin32  = (float*)   alloc(BLC * 4);
  float*    x2_32  = (float*)   alloc(BLC * 4);
  _Float16* xm16   = (_Float16*)alloc(BLC * 2);
  _Float16* fc1_16 = (_Float16*)alloc((size_t)Bn * Ll * 768 * 2);
  _Float16* p_qkv  = (_Float16*)alloc(192 * 576 * 2);
  _Float16* p_proj = (_Float16*)alloc(192 * 192 * 2);
  _Float16* p_fc1  = (_Float16*)alloc(192 * 768 * 2);
  _Float16* p_fc2  = (_Float16*)alloc(768 * 192 * 2);
  _Float16* p_cab1 = (_Float16*)alloc(1728 * 64 * 2);
  _Float16* p_cab2 = (_Float16*)alloc(576 * 192 * 2);
  _Float16* p_hin1 = (_Float16*)alloc(1728 * 192 * 2);
  _Float16* p_hin2 = (_Float16*)alloc(1728 * 192 * 2);
  _Float16* p_hid  = (_Float16*)alloc(192 * 192 * 2);

  auto packW = [&](const float* src, _Float16* dst, int K, int N) {
    int total = (K >> 5) * (N >> 4) * 512;
    pack_b<<<(total + 255) / 256, 256, 0, stream>>>(src, dst, K, N);
  };
  packW(qkv_w,  p_qkv,  192, 576);
  packW(proj_w, p_proj, 192, 192);
  packW(fc1_w,  p_fc1,  192, 768);
  packW(fc2_w,  p_fc2,  768, 192);
  packW(cab1_w, p_cab1, 1728, 64);
  packW(cab2_w, p_cab2, 576, 192);
  packW(hin1_w, p_hin1, 1728, 192);
  packW(hin2_w, p_hin2, 1728, 192);
  packW(hid_w,  p_hid,  192, 192);

  const int M = Bn * Ll;                        // 147456 rows
  const int EW = (int)(BLC / 256);              // elementwise grid
  auto gblk = [](int M_, int N_) { return (((M_ >> 5) * (N_ >> 6)) + 7) / 8; };      // 32x64 waves
  auto cblk = [](int N_) { int Ng = (N_ >= 64) ? (N_ >> 6) : 1;
                           return ((((Bn * Ll) >> 5) * Ng) + 7) / 8; };

  // LN1 -> xn16
  ln_kernel<<<M, 192, 0, stream>>>(x, n1g, n1b, xn16);

  // --- CAB branch ---
  conv_wmma<<<cblk(64), 256, 0, stream>>>(xn16, p_cab1, cab1_b, nullptr, cv1_16, 192, 64, 1);
  conv_wmma<<<cblk(192), 256, 0, stream>>>(cv1_16, p_cab2, cab2_b, cv32, nullptr, 64, 192, 0);
  cab_pool<<<Bn * Cc, 256, 0, stream>>>(cv32, pool32);
  ca_kernel<<<1, 256, 0, stream>>>(pool32, caf1_w, caf1_b, caf2_w, caf2_b, ya32);

  // --- HIN branch ---
  conv_wmma<<<cblk(192), 256, 0, stream>>>(xn16, p_hin1, hin1_b, hv1_32, nullptr, 192, 192, 0);
  hin_stats<<<Bn * 96, 256, 0, stream>>>(hv1_32, hm32, hva32);
  hin_norm<<<EW, 256, 0, stream>>>(hv1_32, hm32, hva32, hing, hinb, hv16);
  conv_wmma<<<cblk(192), 256, 0, stream>>>(hv16, p_hin2, hin2_b, hin32, nullptr, 192, 192, 2);
  gemm_wmma<<<gblk(M, 192), 256, 0, stream>>>(xn16, p_hid, hid_b, hin32, nullptr, M, 192, 192, 0, 1);

  // --- shifted window attention ---
  win_part<<<EW, 256, 0, stream>>>(xn16, wins16);
  gemm_wmma<<<gblk(M, 576), 256, 0, stream>>>(wins16, p_qkv, qkv_b, nullptr, qkv16, M, 576, 192, 0, 0);
  attn_kernel<<<NWIN * NHh, 256, 0, stream>>>(qkv16, rpb, aout16);
  gemm_wmma<<<gblk(M, 192), 256, 0, stream>>>(aout16, p_proj, proj_b, proj32, nullptr, M, 192, 192, 0, 0);

  // residual combine -> x2
  combine_kernel<<<EW, 256, 0, stream>>>(x, proj32, cv32, ya32, hin32, x2_32);

  // --- MLP ---
  ln_kernel<<<M, 192, 0, stream>>>(x2_32, n2g, n2b, xm16);
  gemm_wmma<<<gblk(M, 768), 256, 0, stream>>>(xm16, p_fc1, fc1_b, nullptr, fc1_16, M, 768, 192, 1, 0);
  hipMemcpyAsync(d_out, x2_32, BLC * 4, hipMemcpyDeviceToDevice, stream);
  gemm_wmma<<<gblk(M, 192), 256, 0, stream>>>(fc1_16, p_fc2, fc2_b, (float*)d_out, nullptr, M, 192, 768, 0, 1);
}
